// NeuralTree_17420387352764
// MI455X (gfx1250) — matrix operand order repaired and checked by reading, not compiled, not run on hardware
//
#include <hip/hip_runtime.h>
#include <hip/hip_fp16.h>

// ---------------- problem constants ----------------
static constexpr int FEATURES = 8;
static constexpr int POINTS   = 3;
static constexpr int CLASSES  = 128;
static constexpr int BATCH    = 1024;
static constexpr int NLEAF    = 65536;   // 4^8
static constexpr int HALFK    = 256;     // 4^4

// ---------------- GEMM tiling ----------------
static constexpr int MBLK     = 64;                  // M rows per workgroup (4 WMMA tiles)
static constexpr int KSPLIT   = 32;                  // deterministic split-K factor
static constexpr int KCHUNK   = NLEAF / KSPLIT;      // 2048

// ---------------- workspace layout (bytes) ----------------
static constexpr size_t CT_BYTES = (size_t)CLASSES * NLEAF * 2;  // 16 MB transposed f16 classifier
static constexpr size_t HB_BYTES = (size_t)BATCH * HALFK * 2;    // 512 KB each: hi / lo factors

typedef _Float16 v16h __attribute__((ext_vector_type(16)));
typedef _Float16 v8h  __attribute__((ext_vector_type(8)));
typedef float    v8f  __attribute__((ext_vector_type(8)));

// =====================================================================
// Kernel 1: per-row softmaxes -> hi/lo Kronecker factors (f16)
//   hi[b][i] = prod_{f=0..3} hot[b][f][digit_f(i)]   (i = d0*64+d1*16+d2*4+d3)
//   lo[b][j] = prod_{f=4..7} hot[b][f][digit_f(j)]
// =====================================================================
__global__ void nt_factors(const float* __restrict__ x,
                           const float* __restrict__ beta,
                           _Float16* __restrict__ hi,
                           _Float16* __restrict__ lo) {
  int row = blockIdx.x * blockDim.x + threadIdx.x;
  if (row >= BATCH) return;

  float hf[FEATURES][POINTS + 1];
#pragma unroll
  for (int f = 0; f < FEATURES; ++f) {
    float b0 = beta[f * POINTS + 0];
    float b1 = beta[f * POINTS + 1];
    float b2 = beta[f * POINTS + 2];
    // 3-element ascending sort (min / median / max)
    float s0 = fminf(fminf(b0, b1), b2);
    float s2 = fmaxf(fmaxf(b0, b1), b2);
    float s1 = fmaxf(fminf(b0, b1), fminf(fmaxf(b0, b1), b2));
    float bias[4] = {0.0f, -s0, -s0 - s1, -s0 - s1 - s2};
    float xv = x[row * FEATURES + f];
    float l[4];
    float mx = -3.4e38f;
#pragma unroll
    for (int j = 0; j < 4; ++j) {
      l[j] = (xv * (float)(j + 1) + bias[j]) * 10.0f;  // 1/TEMPERATURE
      mx = fmaxf(mx, l[j]);
    }
    float s = 0.0f;
#pragma unroll
    for (int j = 0; j < 4; ++j) { l[j] = __expf(l[j] - mx); s += l[j]; }
    float inv = 1.0f / s;
#pragma unroll
    for (int j = 0; j < 4; ++j) hf[f][j] = l[j] * inv;
  }

  _Float16* ho = hi + (size_t)row * HALFK;
  _Float16* lw = lo + (size_t)row * HALFK;
#pragma unroll
  for (int d0 = 0; d0 < 4; ++d0) {
    float pA0 = hf[0][d0], pB0 = hf[4][d0];
#pragma unroll
    for (int d1 = 0; d1 < 4; ++d1) {
      float pA1 = pA0 * hf[1][d1], pB1 = pB0 * hf[5][d1];
#pragma unroll
      for (int d2 = 0; d2 < 4; ++d2) {
        float pA2 = pA1 * hf[2][d2], pB2 = pB1 * hf[6][d2];
#pragma unroll
        for (int d3 = 0; d3 < 4; ++d3) {
          int idx = ((d0 * 4 + d1) * 4 + d2) * 4 + d3;
          ho[idx] = (_Float16)(pA2 * hf[3][d3]);
          lw[idx] = (_Float16)(pB2 * hf[7][d3]);
        }
      }
    }
  }
}

// =====================================================================
// Kernel 2: classifier (65536 x 128 f32, row-major) -> Ct (128 x 65536 f16)
// LDS-tiled transpose, coalesced both ways.
// =====================================================================
__global__ void nt_transpose(const float* __restrict__ cls,
                             _Float16* __restrict__ ct) {
  __shared__ _Float16 tile[32][33];  // +1 pad to dodge bank conflicts
  int k0 = blockIdx.x * 32;
  int n0 = blockIdx.y * 32;
  int t = threadIdx.x;          // 256 threads
  int tn = t & 31, tk = t >> 5; // read phase: n fast
#pragma unroll
  for (int r = 0; r < 4; ++r) {
    int kk = tk + r * 8;
    tile[kk][tn] = (_Float16)cls[(size_t)(k0 + kk) * CLASSES + n0 + tn];
  }
  __syncthreads();
  int wk = t & 31, wn = t >> 5; // write phase: k fast
#pragma unroll
  for (int r = 0; r < 4; ++r) {
    int nn = wn + r * 8;
    ct[(size_t)(n0 + nn) * NLEAF + k0 + wk] = tile[wk][nn];
  }
}

// =====================================================================
// Kernel 3: split-K WMMA GEMM.
//   grid = (BATCH/64, KSPLIT), block = 256 threads (8 waves).
//   Workgroup tile: 64(M) x 128(N). Wave w -> M-tile (w>>1), N strip (w&1)*64.
//   Each wave keeps 4 independent accumulators sharing one A fragment:
//   4x less A-synthesis VALU per WMMA and 4-deep XDL ILP.
// =====================================================================
__global__ void __launch_bounds__(256, 1)
nt_gemm(const _Float16* __restrict__ hi,
        const _Float16* __restrict__ lo,
        const _Float16* __restrict__ ct,
        float* __restrict__ partials) {
  __shared__ __align__(16) _Float16 smem[2 * MBLK * HALFK];  // 64 KB: hi rows | lo rows

  int m0 = blockIdx.x * MBLK;
  int ksec = blockIdx.y;
  int t = threadIdx.x;

  // cooperative 64 KB LDS fill (hi/lo rows for this M block; contiguous in global)
  {
    const uint4* gh = (const uint4*)(hi + (size_t)m0 * HALFK);
    const uint4* gl = (const uint4*)(lo + (size_t)m0 * HALFK);
    uint4* sh = (uint4*)smem;
    uint4* sl = (uint4*)(smem + MBLK * HALFK);
    for (int idx = t; idx < (MBLK * HALFK / 8); idx += 256) {
      sh[idx] = gh[idx];
      sl[idx] = gl[idx];
    }
  }
  __syncthreads();

  int wave  = t >> 5;
  int lane  = t & 31;
  int mtile = wave >> 1;                 // 0..3
  int laneg = lane >> 4;                 // lane half selects K sub-range per fragment layout
  int mrow  = mtile * 16 + (lane & 15);  // A row (lanes 0-15 / 16-31 both map M=0..15)
  int nbase = (wave & 1) * 64 + (lane & 15);

  const _Float16* hirow = smem + (size_t)mrow * HALFK;
  const _Float16* lorow = smem + MBLK * HALFK + (size_t)mrow * HALFK;
  const _Float16* bp0 = ct + (size_t)(nbase)      * NLEAF;
  const _Float16* bp1 = ct + (size_t)(nbase + 16) * NLEAF;
  const _Float16* bp2 = ct + (size_t)(nbase + 32) * NLEAF;
  const _Float16* bp3 = ct + (size_t)(nbase + 48) * NLEAF;

  int kbase = ksec * KCHUNK;
  v8f acc0 = {}, acc1 = {}, acc2 = {}, acc3 = {};

  for (int k0 = kbase; k0 < kbase + KCHUNK; k0 += 32) {
    int i0 = k0 >> 8;        // hi index (fixed across aligned 32-wide chunk)
    int j0 = k0 & 255;       // lo base index
    _Float16 hs = hirow[i0];

    // A 16x32 f16 fragment (K = 16*laneg + e consecutive, matching B)
    v8h r1 = *(const v8h*)(lorow + j0 + 16 * laneg);       // ds_load_b128
    v8h r2 = *(const v8h*)(lorow + j0 + 16 * laneg + 8);   // ds_load_b128
    v8h a1 = r1 * hs;
    v8h a2 = r2 * hs;
    v16h a = __builtin_shufflevector(a1, a2, 0, 1, 2, 3, 4, 5, 6, 7,
                                     8, 9, 10, 11, 12, 13, 14, 15);

    // B 32x16 f16 fragments (lane = N column; K = 16*laneg + e consecutive)
    int kb = k0 + 16 * laneg;
    v16h b0 = __builtin_shufflevector(*(const v8h*)(bp0 + kb), *(const v8h*)(bp0 + kb + 8),
                                      0, 1, 2, 3, 4, 5, 6, 7, 8, 9, 10, 11, 12, 13, 14, 15);
    v16h b1 = __builtin_shufflevector(*(const v8h*)(bp1 + kb), *(const v8h*)(bp1 + kb + 8),
                                      0, 1, 2, 3, 4, 5, 6, 7, 8, 9, 10, 11, 12, 13, 14, 15);
    v16h b2 = __builtin_shufflevector(*(const v8h*)(bp2 + kb), *(const v8h*)(bp2 + kb + 8),
                                      0, 1, 2, 3, 4, 5, 6, 7, 8, 9, 10, 11, 12, 13, 14, 15);
    v16h b3 = __builtin_shufflevector(*(const v8h*)(bp3 + kb), *(const v8h*)(bp3 + kb + 8),
                                      0, 1, 2, 3, 4, 5, 6, 7, 8, 9, 10, 11, 12, 13, 14, 15);

    // keep the L2->WGP pipe warm 2 spans (512 k) ahead (uniform branch, EXEC stays full)
    if ((k0 & 255) == 0) {
      __builtin_prefetch(bp0 + k0 + 512, 0, 1);  // global_prefetch_b8
      __builtin_prefetch(bp1 + k0 + 512, 0, 1);
      __builtin_prefetch(bp2 + k0 + 512, 0, 1);
      __builtin_prefetch(bp3 + k0 + 512, 0, 1);
    }

    // 4 independent accumulator chains share one A fragment
    acc0 = __builtin_amdgcn_wmma_f32_16x16x32_f16(false, a, false, b0, (short)0, acc0, false, false);
    acc1 = __builtin_amdgcn_wmma_f32_16x16x32_f16(false, a, false, b1, (short)0, acc1, false, false);
    acc2 = __builtin_amdgcn_wmma_f32_16x16x32_f16(false, a, false, b2, (short)0, acc2, false, false);
    acc3 = __builtin_amdgcn_wmma_f32_16x16x32_f16(false, a, false, b3, (short)0, acc3, false, false);
  }

  // C/D layout: VGPR r -> M = r + 8*laneg ; lane%16 -> N
  float* outp = partials + ((size_t)ksec * BATCH + m0 + mtile * 16) * CLASSES;
#pragma unroll
  for (int r = 0; r < 8; ++r) {
    int rowm = r + 8 * laneg;
    float* rp = outp + (size_t)rowm * CLASSES + nbase;
    rp[0]  = acc0[r];
    rp[16] = acc1[r];
    rp[32] = acc2[r];
    rp[48] = acc3[r];
  }
}

// =====================================================================
// Kernel 4: deterministic fixed-order split-K reduction into d_out.
// =====================================================================
__global__ void nt_reduce(const float* __restrict__ partials,
                          float* __restrict__ out) {
  int idx = blockIdx.x * blockDim.x + threadIdx.x;
  if (idx >= BATCH * CLASSES) return;
  float s = 0.0f;
#pragma unroll
  for (int p = 0; p < KSPLIT; ++p)
    s += partials[(size_t)p * BATCH * CLASSES + idx];
  out[idx] = s;
}

// =====================================================================
extern "C" void kernel_launch(void* const* d_in, const int* in_sizes, int n_in,
                              void* d_out, int out_size, void* d_ws, size_t ws_size,
                              hipStream_t stream) {
  const float* x    = (const float*)d_in[0];   // (1024, 8)
  const float* beta = (const float*)d_in[1];   // (8, 3)
  const float* cls  = (const float*)d_in[2];   // (65536, 128)

  char* ws = (char*)d_ws;
  _Float16* ct       = (_Float16*)(ws);
  _Float16* hi       = (_Float16*)(ws + CT_BYTES);
  _Float16* lo       = (_Float16*)(ws + CT_BYTES + HB_BYTES);
  float*    partials = (float*)   (ws + CT_BYTES + 2 * HB_BYTES);
  float*    out      = (float*)d_out;

  nt_factors<<<dim3(BATCH / 256), 256, 0, stream>>>(x, beta, hi, lo);
  nt_transpose<<<dim3(NLEAF / 32, CLASSES / 32), 256, 0, stream>>>(cls, ct);
  nt_gemm<<<dim3(BATCH / MBLK, KSPLIT), 256, 0, stream>>>(hi, lo, ct, partials);
  nt_reduce<<<dim3((BATCH * CLASSES) / 256), 256, 0, stream>>>(partials, out);
}
